// QLSTMQuantum_65481071400192
// MI455X (gfx1250) — compile-verified, matrix operand order checked by reading, not run on hardware
//
#include <hip/hip_runtime.h>
#include <hip/hip_bf16.h>

typedef _Float16 half16 __attribute__((ext_vector_type(16)));
typedef float    float8 __attribute__((ext_vector_type(8)));

#define T_STEPS 512
#define BATCH   256
#define IN_DIM  512
#define HID     256
#define DTOT    768           // IN_DIM + HID
#define KBLKS   24            // DTOT / 32
#define NTILES  16            // HID / 16

// ---------------- workspace layout (bytes) ----------------
// wpk    : packed f16 weights, 4*24*16*32*16 halves = 1,572,864 B
// biases : 4*256 f32 = 4096 B
// thetas : 4*256 f32 = 4096 B
// cx     : 256*256 f32 = 262,144 B
// hxf16  : 256*256 f16 = 131,072 B
#define WS_WPK    0
#define WS_BIAS   1572864
#define WS_THETA  1576960
#define WS_CX     1581056
#define WS_HX16   1843200

// halves-stride of one K-block (16 n-tiles * 32 lanes * 16 halves)
#define KB_STRIDE_H   8192
// halves-stride of one gate (24 K-blocks)
#define GATE_STRIDE_H (KBLKS * KB_STRIDE_H)

// --------------------------------------------------------------------------
// Pack weights into WMMA B-fragment layout.
// B tile is (K=32) x (N=16).  Lane l (<16): N=l, holds K=0..15 contiguous;
// lane l>=16: N=l-16, holds K=16..31.  16 halves (32B) per lane per fragment.
// --------------------------------------------------------------------------
__global__ void qlstm_pack(const float* __restrict__ Wf, const float* __restrict__ Wi,
                           const float* __restrict__ Wg, const float* __restrict__ Wo,
                           const float* __restrict__ bf, const float* __restrict__ bi,
                           const float* __restrict__ bg, const float* __restrict__ bo,
                           const float* __restrict__ tf, const float* __restrict__ ti,
                           const float* __restrict__ tg, const float* __restrict__ to,
                           _Float16* __restrict__ wpk,
                           float* __restrict__ biases, float* __restrict__ thetas) {
    const int idx = blockIdx.x * blockDim.x + threadIdx.x;
    const float* Ws[4] = {Wf, Wi, Wg, Wo};
    if (idx < 4 * HID * DTOT) {
        const int gate = idx / (HID * DTOT);
        const int rem  = idx % (HID * DTOT);
        const int n    = rem / DTOT;          // output (hidden) index
        const int k    = rem % DTOT;          // reduction index
        const float v  = Ws[gate][n * DTOT + k];
        const int kb = k >> 5, kr = k & 31, nt = n >> 4, nr = n & 15;
        const int lane = nr + ((kr & 16) ? 16 : 0);
        const int j    = kr & 15;
        const size_t dst = (size_t)gate * GATE_STRIDE_H + (size_t)kb * KB_STRIDE_H
                         + (size_t)nt * 512 + (size_t)lane * 16 + j;
        wpk[dst] = (_Float16)v;
    }
    if (idx < 4 * HID) {
        const int gate = idx >> 8, n = idx & 255;
        const float* Bs[4] = {bf, bi, bg, bo};
        const float* Ts[4] = {tf, ti, tg, to};
        biases[idx] = Bs[gate][n];
        thetas[idx] = Ts[gate][n];
    }
}

__global__ void qlstm_init(float* __restrict__ cx, _Float16* __restrict__ hx16) {
    const int idx = blockIdx.x * blockDim.x + threadIdx.x;
    if (idx < BATCH * HID) { cx[idx] = 0.0f; hx16[idx] = (_Float16)0.0f; }
}

// --------------------------------------------------------------------------
// One recurrence step.  grid = 16 blocks (16 batch rows each), 256 threads.
// --------------------------------------------------------------------------
__global__ __launch_bounds__(256)
void qlstm_step(int t, const float* __restrict__ inputs,
                const _Float16* __restrict__ wpk,
                const float* __restrict__ biases, const float* __restrict__ thetas,
                float* __restrict__ cx, _Float16* __restrict__ hx16,
                float* __restrict__ out) {
    __shared__ __align__(32) unsigned char smem[65536];   // aLDS (24KB) then z (64KB), overlaid

    const int tid  = threadIdx.x;
    const int lane = tid & 31;
    const int wave = tid >> 5;
    const int bb   = blockIdx.x;              // batch block (16 rows)

    // ---------------- Phase 1: stage A fragments into LDS ----------------
    // A tile (M=16) x (K=32), lane l: M = l&15, K-half = (l>>4)*16, 16 halves.
    // Fragment slot s = kb*32 + lane; 768 slots total; 3 per thread.
    half16* aLDS = reinterpret_cast<half16*>(smem);
    #pragma unroll
    for (int i = 0; i < 3; ++i) {
        const int s  = tid + i * 256;
        const int kb = s >> 5;
        const int sl = s & 31;
        const int row = bb * 16 + (sl & 15);
        const int kh  = (sl >> 4) * 16;
        half16 frag;
        if (kb < 16) {            // x part: f32 -> f16 convert
            const float4* src = reinterpret_cast<const float4*>(
                inputs + ((size_t)t * BATCH + row) * IN_DIM + kb * 32 + kh);
            #pragma unroll
            for (int q = 0; q < 4; ++q) {
                float4 f = src[q];
                frag[q * 4 + 0] = (_Float16)f.x;
                frag[q * 4 + 1] = (_Float16)f.y;
                frag[q * 4 + 2] = (_Float16)f.z;
                frag[q * 4 + 3] = (_Float16)f.w;
            }
        } else {                  // h part: already f16
            const half16* src = reinterpret_cast<const half16*>(
                hx16 + (size_t)row * HID + (kb - 16) * 32 + kh);
            frag = *src;
        }
        aLDS[s] = frag;
    }
    __syncthreads();

    // ---------------- Phase 2: WMMA GEMM, accumulate in registers --------
    // 64 (gate, ntile) columns over 8 waves -> 8 per wave, 24 K-steps each.
    // Per-pair B pointers hoisted out of the K loop; per K-step all 8 B
    // fragments are loaded as a batch so the scheduler can overlap the
    // loads with the 8 independent wmma chains (partial loadcnt waits).
    const _Float16* bptr[8];
    #pragma unroll
    for (int pi = 0; pi < 8; ++pi) {
        const int p    = wave + pi * 8;
        const int gate = p >> 4;
        const int nt   = p & 15;
        bptr[pi] = wpk + (size_t)gate * GATE_STRIDE_H
                       + (size_t)nt * 512 + (size_t)lane * 16;
    }

    float8 accs[8];
    #pragma unroll
    for (int pi = 0; pi < 8; ++pi) accs[pi] = (float8){0,0,0,0,0,0,0,0};

    for (int kb = 0; kb < KBLKS; ++kb) {
        const half16 a = aLDS[kb * 32 + lane];
        half16 bfrag[8];
        #pragma unroll
        for (int pi = 0; pi < 8; ++pi)
            bfrag[pi] = *reinterpret_cast<const half16*>(bptr[pi] + (size_t)kb * KB_STRIDE_H);
        #pragma unroll
        for (int pi = 0; pi < 8; ++pi)
            accs[pi] = __builtin_amdgcn_wmma_f32_16x16x32_f16(
                false, a, false, bfrag[pi], (short)0, accs[pi], false, false);
    }
    __syncthreads();   // everyone done reading aLDS -> safe to overlay z

    // ---------------- Phase 3: spill z tiles to LDS ----------------------
    // C/D layout: lane<16: N=lane, M=r; lane>=16: N=lane-16, M=8+r.
    typedef float zrow_t[16][HID];
    zrow_t* zsh = reinterpret_cast<zrow_t*>(smem);
    {
        const int ncol = lane & 15;
        const int mof  = (lane >> 4) * 8;
        #pragma unroll
        for (int pi = 0; pi < 8; ++pi) {
            const int p    = wave + pi * 8;
            const int gate = p >> 4;
            const int nt   = p & 15;
            #pragma unroll
            for (int r = 0; r < 8; ++r)
                zsh[gate][mof + r][nt * 16 + ncol] = accs[pi][r];
        }
    }
    __syncthreads();

    // ---------------- Phase 4: cumprod(cos(z + b + theta)) ---------------
    if (tid < 64) {
        const int g = tid >> 4, m = tid & 15;
        float prod = 1.0f;
        for (int n = 0; n < HID; ++n) {
            const float z = zsh[g][m][n] + biases[g * HID + n] + thetas[g * HID + n];
            prod *= __cosf(z);
            zsh[g][m][n] = prod;
        }
    }
    __syncthreads();

    // ---------------- Phase 5: LSTM cell update ---------------------------
    {
        const int m   = tid >> 4;
        const int n0  = (tid & 15) * 16;
        const int row = bb * 16 + m;
        #pragma unroll
        for (int j = 0; j < 16; ++j) {
            const int n = n0 + j;
            const float F = zsh[0][m][n];
            const float I = zsh[1][m][n];
            const float G = zsh[2][m][n];
            const float O = zsh[3][m][n];
            const float f = 1.0f / (1.0f + __expf(-F));
            const float i = 1.0f / (1.0f + __expf(-I));
            const float g = tanhf(G);
            const float o = 1.0f / (1.0f + __expf(-O));
            const size_t si = (size_t)row * HID + n;
            const float c = f * cx[si] + i * g;
            const float h = o * tanhf(c);
            cx[si]   = c;
            hx16[si] = (_Float16)h;
            out[((size_t)t * BATCH + row) * HID + n] = h;                     // stacked[t]
            out[(size_t)T_STEPS * BATCH * HID + si] = h;                      // hx tail
            out[(size_t)T_STEPS * BATCH * HID + (size_t)BATCH * HID + si] = c; // cx tail
        }
    }
}

extern "C" void kernel_launch(void* const* d_in, const int* in_sizes, int n_in,
                              void* d_out, int out_size, void* d_ws, size_t ws_size,
                              hipStream_t stream) {
    const float* inputs = (const float*)d_in[0];
    const float* Wf = (const float*)d_in[1];  const float* bf = (const float*)d_in[2];
    const float* tf = (const float*)d_in[3];
    const float* Wi = (const float*)d_in[4];  const float* bi = (const float*)d_in[5];
    const float* ti = (const float*)d_in[6];
    const float* Wg = (const float*)d_in[7];  const float* bg = (const float*)d_in[8];
    const float* tg = (const float*)d_in[9];
    const float* Wo = (const float*)d_in[10]; const float* bo = (const float*)d_in[11];
    const float* to = (const float*)d_in[12];

    char* ws = (char*)d_ws;
    _Float16* wpk    = (_Float16*)(ws + WS_WPK);
    float*    biases = (float*)(ws + WS_BIAS);
    float*    thetas = (float*)(ws + WS_THETA);
    float*    cx     = (float*)(ws + WS_CX);
    _Float16* hx16   = (_Float16*)(ws + WS_HX16);
    float*    out    = (float*)d_out;

    qlstm_pack<<<(4 * HID * DTOT + 255) / 256, 256, 0, stream>>>(
        Wf, Wi, Wg, Wo, bf, bi, bg, bo, tf, ti, tg, to, wpk, biases, thetas);
    qlstm_init<<<(BATCH * HID + 255) / 256, 256, 0, stream>>>(cx, hx16);

    for (int t = 0; t < T_STEPS; ++t)
        qlstm_step<<<BATCH / 16, 256, 0, stream>>>(t, inputs, wpk, biases, thetas,
                                                   cx, hx16, out);
}
